// CosmosAttention_84035330113993
// MI455X (gfx1250) — matrix operand order repaired and checked by reading, not compiled, not run on hardware
//
#include <hip/hip_runtime.h>

// ---------------------------------------------------------------------------
// CosmosAttention on gfx1250 (MI455X): bf16 WMMA pipeline + async LDS copies
//   K1: QKV GEMM (fp32->bf16 on the fly) + fused RMSNorm + RoPE epilogue
//   K2: Flash attention (async K/V tile streaming, Q.K^T wmma, online softmax,
//       P.V wmma)
//   K3: Output projection GEMM (async bf16 A tiles, fp32 w_o converted) -> f32
// Matrix math: v_wmma_f32_16x16x32_bf16, f32 accumulation.
// Raw bf16 tile movement: global_load_async_to_lds_b128 (ASYNCcnt).
// ---------------------------------------------------------------------------

typedef __attribute__((ext_vector_type(16))) __bf16 v16bf;
typedef __attribute__((ext_vector_type(8)))  float  v8f;

union BF16Frag { v16bf v; unsigned short u[16]; };

__device__ __forceinline__ unsigned short f2bf(float f) {
  union { float f; unsigned u; } x; x.f = f;
  unsigned r = x.u + 0x7FFFu + ((x.u >> 16) & 1u);   // round-to-nearest-even
  return (unsigned short)(r >> 16);
}

__device__ __forceinline__ v8f wmma_bf16(v16bf a, v16bf b, v8f c) {
  return __builtin_amdgcn_wmma_f32_16x16x32_bf16(
      /*neg_a=*/false, a, /*neg_b=*/false, b,
      /*c_mod=*/(short)0, c, /*reuse_a=*/false, /*reuse_b=*/false);
}

// Async copy 16B global -> LDS (GVS mode: 64-bit SGPR base + 32-bit VGPR off).
// lds_off is the byte offset within this workgroup's LDS allocation.
__device__ __forceinline__ void async_g2l_b128(unsigned lds_off,
                                               const void* sbase,
                                               unsigned voff) {
  asm volatile("global_load_async_to_lds_b128 %0, %1, %2"
               :
               : "v"(lds_off), "v"(voff), "s"(sbase)
               : "memory");
}

__device__ __forceinline__ void wait_async0() {
#if __has_builtin(__builtin_amdgcn_s_wait_asynccnt)
  __builtin_amdgcn_s_wait_asynccnt(0);
#else
  asm volatile("s_wait_asynccnt 0x0" ::: "memory");
#endif
}

// A fragment: 16x32 bf16 tile, row-major in LDS (lda elements per row).
// Contiguous k order used here (matches B): lanes 0-15 hold K [0,16); lanes 16-31 hold K [16,32).
__device__ __forceinline__ v16bf load_a_frag(const unsigned short* lds, int lda) {
  int lane = threadIdx.x & 31;
  int r = lane & 15, hi = lane >> 4;
  BF16Frag f;
#pragma unroll
  for (int p = 0; p < 8; ++p) {
    int k = 16 * hi + 2 * p;
    f.u[2 * p]     = lds[r * lda + k];
    f.u[2 * p + 1] = lds[r * lda + k + 1];
  }
  return f.v;
}

// B fragment (32x16) from an [N][K] row-major tile: B[k][n] = lds[n*ldk + k].
// Contiguous k order used here (matches A): lane n holds K in [16*hi, 16*hi+16).
__device__ __forceinline__ v16bf load_b_frag_nk(const unsigned short* lds, int ldk) {
  int lane = threadIdx.x & 31;
  int c = lane & 15, hi = lane >> 4;
  BF16Frag f;
#pragma unroll
  for (int p = 0; p < 8; ++p) {
    int k = 16 * hi + 2 * p;
    f.u[2 * p]     = lds[c * ldk + k];
    f.u[2 * p + 1] = lds[c * ldk + k + 1];
  }
  return f.v;
}

// B fragment (32x16) from a [K][N] row-major tile: B[k][n] = lds[k*ldn + n].
__device__ __forceinline__ v16bf load_b_frag_kn(const unsigned short* lds, int ldn) {
  int lane = threadIdx.x & 31;
  int c = lane & 15, hi = lane >> 4;
  BF16Frag f;
#pragma unroll
  for (int p = 0; p < 8; ++p) {
    int k = 16 * hi + 2 * p;
    f.u[2 * p]     = lds[k * ldn + c];
    f.u[2 * p + 1] = lds[(k + 1) * ldn + c];
  }
  return f.v;
}

// ---------------------------------------------------------------------------
// Kernel 1: QKV projection + RMSNorm + RoPE.
// grid = (M/128=32, inner/128=16 (== head), 3 (q/k/v)), block = 256 (8 waves).
// Tile: BM=128, BN=128, BK=32; wave grid 2x4, wave tile 64x32 (4x2 WMMA tiles).
// (fp32 inputs converted in-flight; no raw-copy opportunity for async here)
// ---------------------------------------------------------------------------
__global__ __launch_bounds__(256) void qkv_kernel(
    const float* __restrict__ x, const float* __restrict__ rope,
    const float* __restrict__ wq, const float* __restrict__ wk,
    const float* __restrict__ wvw, const float* __restrict__ qnw,
    const float* __restrict__ knw,
    unsigned short* __restrict__ qb, unsigned short* __restrict__ kb,
    unsigned short* __restrict__ vb) {
  constexpr int S = 2048, DM = 2048, H = 16;
  extern __shared__ char smem[];
  unsigned short* sA = (unsigned short*)smem;          // 128x32 bf16 (8 KB)
  unsigned short* sB = sA + 128 * 32;                  // 128x32 bf16 (8 KB)
  float* sC = (float*)(smem + 2 * 128 * 32 * 2);       // 128x128 f32 (64 KB)

  const int mBase = blockIdx.x * 128;
  const int h     = blockIdx.y;          // N-tile == head (BN == HD == 128)
  const int nBase = h * 128;
  const int z     = blockIdx.z;          // 0=q, 1=k, 2=v
  const float* w  = (z == 0) ? wq : (z == 1) ? wk : wvw;

  const int tid  = threadIdx.x;
  const int wid  = tid >> 5, lane = tid & 31;
  const int cl   = lane & 15, hi = lane >> 4;
  const int m0   = 64 * (wid & 1), n0 = 32 * (wid >> 1);

  v8f zero = {};
  v8f acc[4][2];
#pragma unroll
  for (int i = 0; i < 4; ++i)
#pragma unroll
    for (int j = 0; j < 2; ++j) acc[i][j] = zero;

  for (int k0 = 0; k0 < DM; k0 += 32) {
    for (int i = tid; i < 128 * 32; i += 256) {
      int r = i >> 5, c = i & 31;
      sA[i] = f2bf(x[(size_t)(mBase + r) * DM + k0 + c]);
      sB[i] = f2bf(w[(size_t)(nBase + r) * DM + k0 + c]);
    }
    __syncthreads();
    v16bf afr[4];
#pragma unroll
    for (int mt = 0; mt < 4; ++mt) afr[mt] = load_a_frag(&sA[(m0 + 16 * mt) * 32], 32);
#pragma unroll
    for (int nt = 0; nt < 2; ++nt) {
      v16bf bfr = load_b_frag_nk(&sB[(n0 + 16 * nt) * 32], 32);
#pragma unroll
      for (int mt = 0; mt < 4; ++mt) acc[mt][nt] = wmma_bf16(afr[mt], bfr, acc[mt][nt]);
    }
    __syncthreads();
  }

  // Stage accumulators to LDS for row-wise epilogue.
#pragma unroll
  for (int mt = 0; mt < 4; ++mt)
#pragma unroll
    for (int nt = 0; nt < 2; ++nt)
#pragma unroll
      for (int v = 0; v < 8; ++v)
        sC[(m0 + 16 * mt + v + 8 * hi) * 128 + (n0 + 16 * nt + cl)] = acc[mt][nt][v];
  __syncthreads();

  if (tid < 128) {
    int gm = mBase + tid;
    int bb = gm / S, s = gm - bb * S;
    float* row = &sC[tid * 128];
    if (z < 2) {
      float ss = 0.f;
      for (int d = 0; d < 128; ++d) ss += row[d] * row[d];
      float rr = rsqrtf(ss * (1.f / 128.f) + 1e-6f);
      const float* nw = (z == 0) ? qnw : knw;
      for (int d = 0; d < 128; ++d) row[d] = row[d] * rr * nw[d];
      unsigned short* dst = ((z == 0) ? qb : kb) + (((size_t)bb * H + h) * S + s) * 128;
      const float* fr = rope + (size_t)s * 64 * 4;   // [S,1,64,2,2]
      for (int j = 0; j < 64; ++j) {
        float t0 = row[j], t1 = row[j + 64];
        const float* f = fr + j * 4;
        dst[j]      = f2bf(f[0] * t0 + f[1] * t1);
        dst[j + 64] = f2bf(f[2] * t0 + f[3] * t1);
      }
    } else {
      unsigned short* dst = vb + (((size_t)bb * H + h) * S + s) * 128;
      for (int d = 0; d < 128; ++d) dst[d] = f2bf(row[d]);
    }
  }
}

// ---------------------------------------------------------------------------
// Kernel 2: flash attention. grid = (S/128=16, B*H=32), block = 256 (8 waves).
// Q/K/V tiles streamed global->LDS via async copy engine (ASYNCcnt).
// Each wave owns 16 query rows; loops over 128-wide K/V tiles.
// LDS map (bytes): sQ @ 0, sK @ 32768, sV @ 65536, sP @ 98304 (128 KB total).
// ---------------------------------------------------------------------------
__global__ __launch_bounds__(256) void attn_kernel(
    const unsigned short* __restrict__ qb, const unsigned short* __restrict__ kb,
    const unsigned short* __restrict__ vb, unsigned short* __restrict__ ao) {
  constexpr int S = 2048, HD = 128, INNER = 2048;
  constexpr unsigned SQ_OFF = 0, SK_OFF = 32768, SV_OFF = 65536;
  extern __shared__ char smem[];
  unsigned short* sQ = (unsigned short*)smem;
  unsigned short* sK = sQ + 128 * 128;
  unsigned short* sV = sK + 128 * 128;
  unsigned short* sP = sV + 128 * 128;

  const int bh = blockIdx.y;                   // bh = b*16 + h
  const int b = bh >> 4, h = bh & 15;
  const size_t headBase = (size_t)bh * S * HD; // elements
  const int qBase = blockIdx.x * 128;

  const int tid = threadIdx.x;
  const int wid = tid >> 5, lane = tid & 31;
  const int cl = lane & 15, hi = lane >> 4;
  const int m0 = 16 * wid;

  const void* qsrc = (const void*)(qb + headBase);
  const void* ksrc = (const void*)(kb + headBase);
  const void* vsrc = (const void*)(vb + headBase);

  // Resident Q tile: 32 KB contiguous, async copy (completion folded into the
  // first iteration's s_wait_asynccnt 0).
  {
    unsigned gq = (unsigned)qBase * HD * 2u;   // byte offset of q tile
#pragma unroll
    for (int j = 0; j < 8; ++j) {
      unsigned idx = (unsigned)(j * 256 + tid) * 16u;
      async_g2l_b128(SQ_OFF + idx, qsrc, gq + idx);
    }
  }

  v8f zero = {};
  v8f o[8]; float mrun[8], lrun[8];
#pragma unroll
  for (int i = 0; i < 8; ++i) { o[i] = zero; mrun[i] = -1e30f; lrun[i] = 0.f; }
  const float scale = 0.08838834764831845f;    // 1/sqrt(128)

  for (int t0 = 0; t0 < S; t0 += 128) {
    __syncthreads();                            // tiles free before overwrite
    {
      unsigned gk = (unsigned)t0 * HD * 2u;     // byte offset of k/v tile
#pragma unroll
      for (int j = 0; j < 8; ++j) {
        unsigned idx = (unsigned)(j * 256 + tid) * 16u;
        async_g2l_b128(SK_OFF + idx, ksrc, gk + idx);
        async_g2l_b128(SV_OFF + idx, vsrc, gk + idx);
      }
    }
    wait_async0();
    __syncthreads();

    // S_ij = Q (16x128) . K^T (128x128)
    v8f sc[8];
#pragma unroll
    for (int nt = 0; nt < 8; ++nt) sc[nt] = zero;
#pragma unroll
    for (int kt = 0; kt < 4; ++kt) {
      v16bf a = load_a_frag(&sQ[m0 * 128 + 32 * kt], 128);
#pragma unroll
      for (int nt = 0; nt < 8; ++nt) {
        v16bf bfr = load_b_frag_nk(&sK[(16 * nt) * 128 + 32 * kt], 128);
        sc[nt] = wmma_bf16(a, bfr, sc[nt]);
      }
    }

    // online softmax; row r lives in v = r%8 across the 16 lanes of half hi = r/8
#pragma unroll
    for (int v = 0; v < 8; ++v) {
      float rmax = -1e30f;
#pragma unroll
      for (int nt = 0; nt < 8; ++nt) { sc[nt][v] *= scale; rmax = fmaxf(rmax, sc[nt][v]); }
#pragma unroll
      for (int msk = 1; msk < 16; msk <<= 1) rmax = fmaxf(rmax, __shfl_xor(rmax, msk, 32));
      float mnew  = fmaxf(mrun[v], rmax);
      float alpha = __expf(mrun[v] - mnew);
      float rsum  = 0.f;
#pragma unroll
      for (int nt = 0; nt < 8; ++nt) { float e = __expf(sc[nt][v] - mnew); sc[nt][v] = e; rsum += e; }
#pragma unroll
      for (int msk = 1; msk < 16; msk <<= 1) rsum += __shfl_xor(rsum, msk, 32);
      lrun[v] = lrun[v] * alpha + rsum;
      mrun[v] = mnew;
#pragma unroll
      for (int dt = 0; dt < 8; ++dt) o[dt][v] *= alpha;
    }

    // P (D-layout) -> wave-private LDS -> re-fragment as A-matrix
#pragma unroll
    for (int nt = 0; nt < 8; ++nt)
#pragma unroll
      for (int v = 0; v < 8; ++v)
        sP[(m0 + v + 8 * hi) * 128 + 16 * nt + cl] = f2bf(sc[nt][v]);

    // O += P (16x128) . V (128x128)
#pragma unroll
    for (int kt = 0; kt < 4; ++kt) {
      v16bf a = load_a_frag(&sP[m0 * 128 + 32 * kt], 128);
#pragma unroll
      for (int dt = 0; dt < 8; ++dt) {
        v16bf bfr = load_b_frag_kn(&sV[(32 * kt) * 128 + 16 * dt], 128);
        o[dt] = wmma_bf16(a, bfr, o[dt]);
      }
    }
  }

  // finalize: O /= l; write [B,S,inner] bf16
#pragma unroll
  for (int v = 0; v < 8; ++v) {
    int sg = qBase + m0 + v + 8 * hi;
    float inv = 1.f / lrun[v];
#pragma unroll
    for (int dt = 0; dt < 8; ++dt) {
      int col = h * HD + 16 * dt + cl;
      ao[((size_t)b * S + sg) * INNER + col] = f2bf(o[dt][v] * inv);
    }
  }
}

// ---------------------------------------------------------------------------
// Kernel 3: out = attn(bf16) @ w_o^T -> fp32. grid = (32,16), block = 256.
// bf16 A tiles streamed via async copies; w_o converted fp32->bf16 in-flight.
// LDS map (bytes): sA @ 0 (8 KB), sB @ 8192 (8 KB).
// ---------------------------------------------------------------------------
__global__ __launch_bounds__(256) void out_kernel(
    const unsigned short* __restrict__ a, const float* __restrict__ wo,
    float* __restrict__ out) {
  constexpr int K = 2048, N = 2048;
  extern __shared__ char smem[];
  unsigned short* sA = (unsigned short*)smem;          // 128x32 bf16
  unsigned short* sB = sA + 128 * 32;                  // 128x32 bf16
  const int mBase = blockIdx.x * 128;
  const int nBase = blockIdx.y * 128;
  const int tid = threadIdx.x;
  const int wid = tid >> 5, lane = tid & 31;
  const int cl = lane & 15, hi = lane >> 4;
  const int m0 = 64 * (wid & 1), n0 = 32 * (wid >> 1);

  v8f zero = {};
  v8f acc[4][2];
#pragma unroll
  for (int i = 0; i < 4; ++i)
#pragma unroll
    for (int j = 0; j < 2; ++j) acc[i][j] = zero;

  const void* asrc = (const void*)(a + (size_t)mBase * K);   // tile row base
  for (int k0 = 0; k0 < K; k0 += 32) {
    // A tile: 128 rows x 64 bytes, async; row stride in global = 4096 bytes.
#pragma unroll
    for (int j = 0; j < 2; ++j) {
      int i = j * 256 + tid;          // 512 chunks of 16B
      int r = i >> 2, c = i & 3;
      async_g2l_b128((unsigned)(r * 64 + c * 16), asrc,
                     (unsigned)(r * 4096 + k0 * 2 + c * 16));
    }
    // B tile: fp32 -> bf16 conversion path.
    for (int i = tid; i < 128 * 32; i += 256) {
      int r = i >> 5, c = i & 31;
      sB[i] = f2bf(wo[(size_t)(nBase + r) * K + k0 + c]);
    }
    wait_async0();
    __syncthreads();
    v16bf afr[4];
#pragma unroll
    for (int mt = 0; mt < 4; ++mt) afr[mt] = load_a_frag(&sA[(m0 + 16 * mt) * 32], 32);
#pragma unroll
    for (int nt = 0; nt < 2; ++nt) {
      v16bf bfr = load_b_frag_nk(&sB[(n0 + 16 * nt) * 32], 32);
#pragma unroll
      for (int mt = 0; mt < 4; ++mt) acc[mt][nt] = wmma_bf16(afr[mt], bfr, acc[mt][nt]);
    }
    __syncthreads();
  }

#pragma unroll
  for (int mt = 0; mt < 4; ++mt)
#pragma unroll
    for (int nt = 0; nt < 2; ++nt)
#pragma unroll
      for (int v = 0; v < 8; ++v)
        out[(size_t)(mBase + m0 + 16 * mt + v + 8 * hi) * N + (nBase + n0 + 16 * nt + cl)] =
            acc[mt][nt][v];
}

// ---------------------------------------------------------------------------
extern "C" void kernel_launch(void* const* d_in, const int* in_sizes, int n_in,
                              void* d_out, int out_size, void* d_ws, size_t ws_size,
                              hipStream_t stream) {
  const float* x    = (const float*)d_in[0];
  const float* rope = (const float*)d_in[1];
  const float* wq   = (const float*)d_in[2];
  const float* wk   = (const float*)d_in[3];
  const float* wvw  = (const float*)d_in[4];
  const float* wo   = (const float*)d_in[5];
  const float* qnw  = (const float*)d_in[6];
  const float* knw  = (const float*)d_in[7];
  float* out = (float*)d_out;

  // Workspace: 4 x 16 MiB bf16 buffers (q, k, v in [B,H,S,D]; attn in [B,S,inner])
  const size_t BUF = (size_t)2 * 16 * 2048 * 128 * sizeof(unsigned short);
  unsigned short* qb = (unsigned short*)d_ws;
  unsigned short* kb = (unsigned short*)((char*)d_ws + 1 * BUF);
  unsigned short* vb = (unsigned short*)((char*)d_ws + 2 * BUF);
  unsigned short* ao = (unsigned short*)((char*)d_ws + 3 * BUF);

  dim3 blk(256, 1, 1);

  dim3 g1(32, 16, 3);   // M-tiles, heads, q/k/v
  qkv_kernel<<<g1, blk, 80 * 1024, stream>>>(x, rope, wq, wk, wvw, qnw, knw, qb, kb, vb);

  dim3 g2(16, 32, 1);   // q-row tiles, B*H
  attn_kernel<<<g2, blk, 128 * 1024, stream>>>(qb, kb, vb, ao);

  dim3 g3(32, 16, 1);   // M-tiles, N-tiles
  out_kernel<<<g3, blk, 16 * 1024, stream>>>(ao, wo, out);
}
